// LinearAttn_75325136437308
// MI455X (gfx1250) — matrix operand order repaired and checked by reading, not compiled, not run on hardware
//
#include <hip/hip_runtime.h>
#include <hip/hip_bf16.h>

// ---------------------------------------------------------------------------
// LinearAttn (Taylor feature map) fused pipeline for gfx1250 (MI455X, wave32)
//
//   K0  pack W_qkv / W_out -> bf16, B-fragment order [K/16][N][16]
//   Kz  zero kv / ksum accumulators
//   K1  qkv = query @ W_qkv            (v_wmma_f32_16x16x32_bf16)
//   K2  kv[b,h] = phi(k)^T v, ksum     (v_wmma_f32_16x16x32_bf16 + f32 atomics)
//   K3  out = (phi(q)kv/den) @ W_out   (scalar f32 + v_wmma_f32_16x16x32_bf16)
//
// WMMA lane layouts (CDNA5 ISA, wave32):
//   A 16x32 bf16 : M = lane&15; halves j=0..7 -> K = (lane>>4)*8 + j,
//                  j=8..15 -> K = 16 + (lane>>4)*8 + (j-8)   (2 contiguous runs)
//   B 32x16 bf16 : N = lane&15; this code: half j -> K = (lane>>4)*16 + j, A uses the same order
//   C/D 16x16 f32: VGPR r -> M = (lane>>4)*8 + r, N = lane&15
// ---------------------------------------------------------------------------

typedef __attribute__((ext_vector_type(16))) __bf16 v16bf;
typedef __attribute__((ext_vector_type(8)))  float  v8f;

#define BATCH   8
#define SEQ     4096
#define DIM     512
#define HEADS   8
#define DH      16
#define QKVN    384        // 3*HEADS*DH
#define DREAL   273        // 1 + 16 + 256 taylor features
#define DPAD    288        // padded to 18 * 16
#define SCALE   0.25f      // DH^-0.5
#define EPSV    1e-5f

union BF16Frag {
    v16bf v;
    unsigned short u[16];
};

__device__ __forceinline__ unsigned short f2bf(float f) {
    union { float f; unsigned int u; } c; c.f = f;
    unsigned int x = c.u;
    unsigned int r = x + 0x7FFFu + ((x >> 16) & 1u);   // round-to-nearest-even
    return (unsigned short)(r >> 16);
}

__device__ __forceinline__ v8f wmma_bf16(v16bf a, v16bf b, v8f c) {
    // (neg_a, A, neg_b, B, c_mod, C, reuse_a, reuse_b)
    return __builtin_amdgcn_wmma_f32_16x16x32_bf16(false, a, false, b,
                                                   (short)0, c, false, false);
}

// taylor feature d of a 16-vector x: [1, x0..x15, sqrt(.5)*x_i*x_j (i-major)]
__device__ __forceinline__ float featf(const float* x16, int d) {
    if (d == 0) return 1.0f;
    if (d < 17) return x16[d - 1];
    if (d < DREAL) {
        int p = d - 17;
        return 0.70710678118654752f * x16[p >> 4] * x16[p & 15];
    }
    return 0.0f;   // padding rows 273..287
}

// ---------------------------------------------------------------------------
// K0: pack weights to bf16 fragment order. wpack[kk][n][j] = W_qkv[kk*16+j][n]
//     (kk<32, n<384), wopack[kk][n][j] = W_out[kk*16+j][n] (kk<8, n<512).
// ---------------------------------------------------------------------------
__global__ __launch_bounds__(256) void la_pack(const float* __restrict__ wqkv,
                                               const float* __restrict__ wout,
                                               unsigned short* __restrict__ wpack,
                                               unsigned short* __restrict__ wopack) {
    int idx = blockIdx.x * 256 + threadIdx.x;
    if (idx < 32 * QKVN * 16) {
        int j  = idx & 15;
        int n  = (idx >> 4) % QKVN;
        int kk = idx / (QKVN * 16);
        wpack[idx] = f2bf(wqkv[(size_t)(kk * 16 + j) * QKVN + n]);
    }
    if (idx < 8 * DIM * 16) {
        int j  = idx & 15;
        int n  = (idx >> 4) % DIM;
        int kk = idx / (DIM * 16);
        wopack[idx] = f2bf(wout[(size_t)(kk * 16 + j) * DIM + n]);
    }
}

__global__ __launch_bounds__(256) void la_zero(float* __restrict__ kv,
                                               float* __restrict__ ksum) {
    int idx = blockIdx.x * 256 + threadIdx.x;
    if (idx < BATCH * HEADS * DPAD * DH) kv[idx] = 0.0f;
    if (idx < BATCH * HEADS * DPAD)      ksum[idx] = 0.0f;
}

// ---------------------------------------------------------------------------
// K1: qkv[32768][384] = query[32768][512] @ W_qkv[512][384]
// block = 256 thr (8 waves); block tile 128(M) x 64(N); wave tile 16 x 64.
// A fragments straight from global (two 32B runs/lane), B from packed bf16.
// ---------------------------------------------------------------------------
__global__ __launch_bounds__(256) void la_qkv(const float* __restrict__ query,
                                              const unsigned short* __restrict__ wpack,
                                              float* __restrict__ qkv) {
    int blk  = blockIdx.x;
    int nblk = blk % (QKVN / 64);          // 6 column blocks
    int mblk = blk / (QKVN / 64);          // 256 row blocks
    int tid  = threadIdx.x;
    int wave = tid >> 5, lane = tid & 31;
    int lh   = lane >> 4, lm = lane & 15;

    int rowA  = mblk * 128 + wave * 16 + lm;
    int nbase = nblk * 64;
    const float* arow = query + (size_t)rowA * DIM;

    v8f acc[4] = {};
    for (int ks = 0; ks < DIM / 32; ++ks) {
        BF16Frag af;
        int k0 = ks * 32 + lh * 16;
        #pragma unroll
        for (int j = 0; j < 8; ++j) af.u[j]     = f2bf(arow[k0 + j]);
        #pragma unroll
        for (int j = 0; j < 8; ++j) af.u[8 + j] = f2bf(arow[k0 + 8 + j]);
        #pragma unroll
        for (int t = 0; t < 4; ++t) {
            int n = nbase + t * 16 + lm;
            const unsigned short* bp = wpack + ((size_t)(ks * 2 + lh) * QKVN + n) * 16;
            BF16Frag bf;
            #pragma unroll
            for (int j = 0; j < 16; ++j) bf.u[j] = bp[j];
            acc[t] = wmma_bf16(af.v, bf.v, acc[t]);
        }
    }
    #pragma unroll
    for (int t = 0; t < 4; ++t) {
        int col = nbase + t * 16 + lm;
        #pragma unroll
        for (int r = 0; r < 8; ++r) {
            int row = mblk * 128 + wave * 16 + lh * 8 + r;
            qkv[(size_t)row * QKVN + col] = acc[t][r];
        }
    }
}

// ---------------------------------------------------------------------------
// K2: kv[b,h][288][16] += phi(k)^T v  over a 512-token segment; ksum likewise.
// 512 blocks = 64 (b,h) pairs x 8 segments. Per 32-token chunk, stage k/v head
// values in LDS, build phi A-fragments on the fly, WMMA-accumulate, atomics out.
// ---------------------------------------------------------------------------
__global__ __launch_bounds__(256) void la_kv(const float* __restrict__ qkv,
                                             const unsigned char* __restrict__ amask,
                                             float* __restrict__ kv,
                                             float* __restrict__ ksum) {
    __shared__ float kbuf[32][16];
    __shared__ float vbuf[32][16];

    int blk = blockIdx.x;
    int seg = blk & 7;
    int bh  = blk >> 3;
    int b   = bh >> 3, h = bh & 7;
    int tid = threadIdx.x;
    int wave = tid >> 5, lane = tid & 31;
    int lh = lane >> 4, lm = lane & 15;

    int ntile = (wave < 2) ? 3 : 2;         // 18 d-tiles over 8 waves
    v8f acc[3] = {};
    float ks0 = 0.0f, ks1 = 0.0f;           // ksum dims tid and tid+256
    int tok0 = seg * 512;

    for (int chunk = 0; chunk < 16; ++chunk) {
        int t0 = tok0 + chunk * 32;
        for (int idx = tid; idx < 512; idx += 256) {
            int t = idx >> 4, i = idx & 15;
            int token = t0 + t;
            float m = amask[(size_t)b * SEQ + token] ? 1.0f : 0.0f;
            const float* base = qkv + (size_t)(b * SEQ + token) * QKVN + h * DH + i;
            kbuf[t][i] = base[128] * m;     // k slice
            vbuf[t][i] = base[256] * m;     // v slice
        }
        __syncthreads();

        // ksum partials (register-resident per owner thread)
        {
            float s = 0.0f;
            for (int t = 0; t < 32; ++t) s += featf(&kbuf[t][0], tid);
            ks0 += s;
        }
        if (tid < DPAD - 256) {
            float s = 0.0f;
            int d = tid + 256;
            for (int t = 0; t < 32; ++t) s += featf(&kbuf[t][0], d);
            ks1 += s;
        }

        // B fragment (v), shared by all d-tiles of this wave
        BF16Frag bf;
        #pragma unroll
        for (int j = 0; j < 16; ++j) bf.u[j] = f2bf(vbuf[lh * 16 + j][lm]);

        for (int s = 0; s < ntile; ++s) {
            int dtile = wave + s * 8;
            int d = dtile * 16 + lm;
            BF16Frag af;
            #pragma unroll
            for (int j = 0; j < 16; ++j) {
                int t = lh * 16 + j;
                af.u[j] = f2bf(featf(&kbuf[t][0], d));
            }
            acc[s] = wmma_bf16(af.v, bf.v, acc[s]);
        }
        __syncthreads();
    }

    for (int s = 0; s < ntile; ++s) {
        int dtile = wave + s * 8;
        #pragma unroll
        for (int r = 0; r < 8; ++r) {
            int d = dtile * 16 + lh * 8 + r;
            atomicAdd(&kv[((size_t)bh * DPAD + d) * DH + lm], acc[s][r]);
        }
    }
    atomicAdd(&ksum[(size_t)bh * DPAD + tid], ks0);
    if (tid < DPAD - 256) atomicAdd(&ksum[(size_t)bh * DPAD + 256 + tid], ks1);
}

// ---------------------------------------------------------------------------
// K3: per 64-token block of one batch: headout = phi(q)*kv/den (scalar f32),
// then out(64x512) = headout(64x128 bf16) @ W_out via WMMA.
// 512 blocks = 8 batches x 64 token-blocks.
// ---------------------------------------------------------------------------
__global__ __launch_bounds__(256) void la_out(const float* __restrict__ qkv,
                                              const float* __restrict__ kv,
                                              const float* __restrict__ ksum,
                                              const unsigned short* __restrict__ wopack,
                                              float* __restrict__ out) {
    __shared__ float          kvh[DPAD * DH];    // 18 KB
    __shared__ float          ksh[DPAD];
    __shared__ float          qh[64 * DH];       // 4 KB
    __shared__ unsigned short hout[64 * 128];    // 16 KB bf16

    int blk  = blockIdx.x;
    int b    = blk >> 6;
    int tok0 = (blk & 63) * 64;
    int tid  = threadIdx.x;
    int wave = tid >> 5, lane = tid & 31;
    int lh = lane >> 4, lm = lane & 15;

    for (int h = 0; h < HEADS; ++h) {
        int bh = b * HEADS + h;
        for (int idx = tid; idx < DPAD * DH; idx += 256)
            kvh[idx] = kv[(size_t)bh * DPAD * DH + idx];
        for (int idx = tid; idx < DPAD; idx += 256)
            ksh[idx] = ksum[(size_t)bh * DPAD + idx];
        for (int idx = tid; idx < 64 * DH; idx += 256) {
            int t = idx >> 4, i = idx & 15;
            qh[idx] = qkv[(size_t)(b * SEQ + tok0 + t) * QKVN + h * DH + i] * SCALE;
        }
        __syncthreads();

        int token = tid & 63;
        int e0    = (tid >> 6) * 4;              // 4 threads split 16 outputs
        const float* qv = &qh[token * DH];
        float den = 0.0f, a0 = 0.0f, a1 = 0.0f, a2 = 0.0f, a3 = 0.0f;
        for (int d = 0; d < DREAL; ++d) {
            float f = featf(qv, d);
            const float* kr = &kvh[d * DH + e0];
            den += f * ksh[d];
            a0 += f * kr[0]; a1 += f * kr[1]; a2 += f * kr[2]; a3 += f * kr[3];
        }
        float inv = 1.0f / fmaxf(den, EPSV);
        unsigned short* hp = &hout[token * 128 + h * DH + e0];
        hp[0] = f2bf(a0 * inv); hp[1] = f2bf(a1 * inv);
        hp[2] = f2bf(a2 * inv); hp[3] = f2bf(a3 * inv);
        __syncthreads();
    }

    // output projection: 4 M-tiles x 32 N-tiles over 8 waves
    int mt = wave & 3;
    int ng = wave >> 2;
    for (int nti = 0; nti < 16; ++nti) {
        int nt = ng * 16 + nti;
        v8f acc = {};
        for (int ks = 0; ks < 4; ++ks) {
            BF16Frag af;
            const unsigned short* ap = &hout[(mt * 16 + lm) * 128 + ks * 32 + lh * 16];
            #pragma unroll
            for (int j = 0; j < 8; ++j) af.u[j]     = ap[j];
            #pragma unroll
            for (int j = 0; j < 8; ++j) af.u[8 + j] = ap[8 + j];
            BF16Frag bf;
            const unsigned short* bp = wopack + ((size_t)(ks * 2 + lh) * DIM + nt * 16 + lm) * 16;
            #pragma unroll
            for (int j = 0; j < 16; ++j) bf.u[j] = bp[j];
            acc = wmma_bf16(af.v, bf.v, acc);
        }
        #pragma unroll
        for (int r = 0; r < 8; ++r) {
            int row = tok0 + mt * 16 + lh * 8 + r;
            out[(size_t)(b * SEQ + row) * DIM + nt * 16 + lm] = acc[r];
        }
    }
}

// ---------------------------------------------------------------------------
extern "C" void kernel_launch(void* const* d_in, const int* in_sizes, int n_in,
                              void* d_out, int out_size, void* d_ws, size_t ws_size,
                              hipStream_t stream) {
    (void)in_sizes; (void)n_in; (void)out_size; (void)ws_size;

    const float*         query  = (const float*)d_in[0];
    const unsigned char* amask  = (const unsigned char*)d_in[3];   // bool (b,n)
    const float*         w_qkv  = (const float*)d_in[6];
    const float*         w_out  = (const float*)d_in[7];
    float*               out    = (float*)d_out;

    // workspace carve-up (~52 MB total)
    float* qkv_ws  = (float*)d_ws;                                  // 32768*384 f32
    float* kv_ws   = qkv_ws + (size_t)BATCH * SEQ * QKVN;           // 64*288*16 f32
    float* ksum_ws = kv_ws + (size_t)BATCH * HEADS * DPAD * DH;     // 64*288 f32
    unsigned short* wpack  = (unsigned short*)(ksum_ws + BATCH * HEADS * DPAD);
    unsigned short* wopack = wpack + (size_t)32 * QKVN * 16;        // 512*384 bf16

    la_pack<<<(32 * QKVN * 16 + 255) / 256, 256, 0, stream>>>(w_qkv, w_out, wpack, wopack);
    la_zero<<<(BATCH * HEADS * DPAD * DH + 255) / 256, 256, 0, stream>>>(kv_ws, ksum_ws);
    la_qkv<<<(BATCH * SEQ / 128) * (QKVN / 64), 256, 0, stream>>>(query, wpack, qkv_ws);
    la_kv<<<BATCH * HEADS * 8, 256, 0, stream>>>(qkv_ws, amask, kv_ws, ksum_ws);
    la_out<<<BATCH * (SEQ / 64), 256, 0, stream>>>(qkv_ws, kv_ws, ksum_ws, wopack, out);
}
